// TokenAwareFFN_59399397703836
// MI455X (gfx1250) — matrix operand order repaired and checked by reading, not compiled, not run on hardware
//
#include <hip/hip_runtime.h>
#include <hip/hip_bf16.h>

#define N_TOK 32768   // B*S = 8*4096
#define Dm    512
#define Fm    2048
#define Tm    4
#define FC    256     // F-chunk width for fused GEMM1->GEMM2
#define MT    32      // token rows per workgroup tile (2 WMMA M-tiles)
#define LSTR  520     // LDS row stride for X (ushorts), padded (520*2B = 16B mult)
#define HSTR  264     // LDS row stride for H chunk (ushorts), padded

typedef __attribute__((ext_vector_type(16))) __bf16 v16bf;
typedef __attribute__((ext_vector_type(8)))  float  v8f;

union Frag {
    unsigned int u[8];
    uint4        q[2];
    v16bf        v;
};

__device__ inline unsigned short f2bf(float f) {
    unsigned int u = __float_as_uint(f);
    return (unsigned short)((u + 0x7fffu + ((u >> 16) & 1u)) >> 16); // RNE
}

__device__ inline v8f vzero8() {
    v8f z = {0.f, 0.f, 0.f, 0.f, 0.f, 0.f, 0.f, 0.f};
    return z;
}

// WGP-scope prefetch (scope 0 fills all cache levels) with an immediate byte
// offset so the compiler can feed it the same walking base register the loads
// use instead of materializing extra 64-bit address adds.
template <int BOFF>
__device__ inline void prefetch_wgp(const void* p) {
    asm volatile("global_prefetch_b8 %0, off offset:%c1" ::"v"(p), "i"(BOFF));
}

// Branch-free GELU (tanh form): h * sigmoid(1.59576912*(h + 0.044715 h^3)).
// Uses v_exp_f32 / v_rcp_f32 TRANS ops; error ~1e-3, below bf16 matmul noise.
__device__ inline float gelu_fast(float h) {
    const float u = 0.7978845608028654f * (h + 0.044715f * h * h * h);
    // sigmoid(2u) = 1 / (1 + exp(-2u)); exp(-2u) = exp2(u * -2*log2(e))
    const float e = __builtin_amdgcn_exp2f(u * -2.8853900817779268f);
    return h * __builtin_amdgcn_rcpf(1.0f + e);
}

// A-matrix fragment (16x32 bf16), from an LDS tile with
// row stride `stride` ushorts. lane m=lane&15; lanes 0-15: K=k0+[0..15];
// lanes 16-31: K=k0+[16..31] (matches load_b).
__device__ inline v16bf load_a(const unsigned short* base, int lane, int k0,
                               int stride) {
    const int m  = lane & 15;
    const int b0 = (lane >> 4) << 4; // 0 or 16
    const uint4* p = reinterpret_cast<const uint4*>(base + m * stride + k0 + b0);
    Frag a;
    a.q[0] = p[0];   // K = b0 .. b0+7
    a.q[1] = p[1];   // K = b0+8 .. b0+15
    return a.v;
}

// B-matrix fragment (32x16 bf16): lane holds column n=lane&15, 16 contiguous K
// values starting at K = (lane<16 ? 0 : 16). Caller passes lane's 32B address.
__device__ inline v16bf load_b(const unsigned short* g) {
    const uint4* p = reinterpret_cast<const uint4*>(g);
    Frag b;
    b.q[0] = p[0];
    b.q[1] = p[1];
    return b.v;
}

// ---------------------------------------------------------------------------
// Kernel 1: stable counting sort of tokens by type (deterministic).
// meta[0..3] = per-type counts, meta[4..8] = padded exclusive offsets (mult MT)
// idxs[padded pos] = token index, -1 for pad slots.
// ---------------------------------------------------------------------------
__global__ __launch_bounds__(1024)
void build_groups(const int* __restrict__ tt, int* __restrict__ meta,
                  int* __restrict__ idxs) {
    __shared__ int cnt[1024 * 4];
    __shared__ int poffs[5];
    const int tid  = threadIdx.x;
    const int base = tid * (N_TOK / 1024); // 32 tokens per thread, contiguous

    int c[4] = {0, 0, 0, 0};
    for (int i = 0; i < N_TOK / 1024; ++i) c[tt[base + i] & 3]++;
#pragma unroll
    for (int t = 0; t < 4; ++t) cnt[tid * 4 + t] = c[t];
    __syncthreads();

    if (tid == 0) {
        int run[4] = {0, 0, 0, 0};
        for (int i = 0; i < 1024; ++i) {
#pragma unroll
            for (int t = 0; t < 4; ++t) {
                int v = cnt[i * 4 + t];
                cnt[i * 4 + t] = run[t];  // exclusive base per (thread, type)
                run[t] += v;
            }
        }
        int p = 0;
#pragma unroll
        for (int t = 0; t < 4; ++t) {
            meta[t]  = run[t];
            poffs[t] = p;
            p += ((run[t] + MT - 1) / MT) * MT; // pad group to tile multiple
        }
        poffs[4] = p;
#pragma unroll
        for (int t = 0; t < 5; ++t) meta[4 + t] = poffs[t];
    }
    __syncthreads();

    const int ptot = poffs[4];
    for (int i = tid; i < ptot; i += 1024) idxs[i] = -1;
    __syncthreads();

    int b[4];
#pragma unroll
    for (int t = 0; t < 4; ++t) b[t] = cnt[tid * 4 + t];
    for (int i = 0; i < N_TOK / 1024; ++i) {
        const int tok = base + i;
        const int ty  = tt[tok] & 3;
        idxs[poffs[ty] + b[ty]] = tok;
        b[ty]++;
    }
}

// ---------------------------------------------------------------------------
// Kernel 2a/2b: convert + transpose weights to bf16 (16 MB total -> lives in
// the 192 MB L2 for the whole FFN pass).
// W1 [T][D][F] -> W1T [T][F][D] ; W2 [T][F][D] -> W2T [T][D][F]
// ---------------------------------------------------------------------------
__global__ __launch_bounds__(256)
void convert_w1(const float* __restrict__ W1, unsigned short* __restrict__ W1T) {
    const int idx = blockIdx.x * 256 + threadIdx.x; // over T*F*D (output order)
    const int fd = Fm * Dm;
    const int t = idx / fd;
    const int r = idx - t * fd;
    const int f = r / Dm;
    const int d = r - f * Dm;
    W1T[idx] = f2bf(W1[(size_t)t * Dm * Fm + (size_t)d * Fm + f]);
}

__global__ __launch_bounds__(256)
void convert_w2(const float* __restrict__ W2, unsigned short* __restrict__ W2T) {
    const int idx = blockIdx.x * 256 + threadIdx.x; // over T*D*F (output order)
    const int df = Dm * Fm;
    const int t = idx / df;
    const int r = idx - t * df;
    const int d = r / Fm;
    const int f = r - d * Fm;
    W2T[idx] = f2bf(W2[(size_t)t * Fm * Dm + (size_t)f * Dm + d]);
}

// ---------------------------------------------------------------------------
// Kernel 3: fused grouped FFN. One 32-token tile (single type) per WG.
// 8 wave32 per WG; wave w: M-tile = w&1, N-group = w>>1.
// GEMM1 (D=512 reduce) -> gelu -> LDS -> GEMM2 (FC=256 reduce), F in 8 chunks.
// ---------------------------------------------------------------------------
__global__ __launch_bounds__(256)
void ffn_kernel(const float* __restrict__ x, const int* __restrict__ meta,
                const int* __restrict__ idxs,
                const unsigned short* __restrict__ W1T,
                const float* __restrict__ b1,
                const unsigned short* __restrict__ W2T,
                const float* __restrict__ b2, float* __restrict__ out) {
    __shared__ unsigned short Xl[MT * LSTR]; // X tile, bf16, row-major  (~33 KB)
    __shared__ unsigned short Hl[MT * HSTR]; // gelu(H) chunk, bf16     (~17 KB)

    const int tid  = threadIdx.x;
    const int wave = tid >> 5;
    const int lane = tid & 31;

    int poff[5];
#pragma unroll
    for (int i = 0; i < 5; ++i) poff[i] = meta[4 + i];
    const int r0 = blockIdx.x * MT;
    if (r0 >= poff[4]) return;

    int t = 0;
    while (t < 3 && r0 >= poff[t + 1]) t++;
    const int cnt       = meta[t];
    const int validRows = min(MT, poff[t] + cnt - r0);

    // --- stage X tile into LDS as bf16 (8 threads per row, float4 loads) ---
    {
        const int m   = tid >> 3; // row 0..31
        const int seg = tid & 7;  // 64 floats per segment
        const int tok = (m < validRows) ? idxs[r0 + m] : -1;
        unsigned short* dst = Xl + m * LSTR + seg * 64;
        if (tok >= 0) {
            const float4* src =
                reinterpret_cast<const float4*>(x + (size_t)tok * Dm + seg * 64);
#pragma unroll
            for (int i = 0; i < 16; ++i) {
                float4 v = src[i];
                dst[i * 4 + 0] = f2bf(v.x);
                dst[i * 4 + 1] = f2bf(v.y);
                dst[i * 4 + 2] = f2bf(v.z);
                dst[i * 4 + 3] = f2bf(v.w);
            }
        } else {
#pragma unroll
            for (int i = 0; i < 64; ++i) dst[i] = 0;
        }
    }
    __syncthreads();

    const int mt    = wave & 1;   // which 16-row M-tile
    const int grp   = wave >> 1;  // N-group 0..3
    const int nlane = lane & 15;
    const int khalf = (lane >> 4) << 4; // B-fragment K half: 0 or 16
    const int mbase = (lane >> 4) << 3; // C-fragment M half: 0 or 8

    const unsigned short* XlA = Xl + mt * 16 * LSTR;
    const unsigned short* HlA = Hl + mt * 16 * HSTR;

    v8f cy[8];
#pragma unroll
    for (int j = 0; j < 8; ++j) cy[j] = vzero8();

    for (int fc = 0; fc < Fm / FC; ++fc) {
        v8f ch[4];
#pragma unroll
        for (int j = 0; j < 4; ++j) ch[j] = vzero8();

        // ---- GEMM1: H[32,FC] = X[32,D] * W1T, reduce over D ----
        const unsigned short* w1p =
            W1T + ((size_t)t * Fm + fc * FC + grp * 64 + nlane) * Dm + khalf;
        for (int k0 = 0; k0 < Dm; k0 += 32) {
            const unsigned short* gb = w1p + k0; // per-j stride = 16*Dm*2 bytes
            // prefetch 4 k-iterations ahead on all four j-streams (imm offsets)
            prefetch_wgp<0 * 16384 + 256>(gb);
            prefetch_wgp<1 * 16384 + 256>(gb);
            prefetch_wgp<2 * 16384 + 256>(gb);
            prefetch_wgp<3 * 16384 + 256>(gb);
            v16bf a = load_a(XlA, lane, k0, LSTR);
            v16bf bfr[4];
#pragma unroll
            for (int j = 0; j < 4; ++j) bfr[j] = load_b(gb + (size_t)j * 16 * Dm);
#pragma unroll
            for (int j = 0; j < 4; ++j)
                ch[j] = __builtin_amdgcn_wmma_f32_16x16x32_bf16(
                    false, a, false, bfr[j], (short)0, ch[j], false, false);
        }

        __syncthreads(); // prior chunk's GEMM2 reads of Hl are done

        // ---- bias + gelu + write bf16 H chunk to LDS ----
#pragma unroll
        for (int j = 0; j < 4; ++j) {
            const int nloc = grp * 64 + j * 16 + nlane;
            const float bias = b1[t * Fm + fc * FC + nloc];
#pragma unroll
            for (int r = 0; r < 8; ++r) {
                const int m = mt * 16 + r + mbase;
                Hl[m * HSTR + nloc] = f2bf(gelu_fast(ch[j][r] + bias));
            }
        }
        __syncthreads();

        // ---- GEMM2: Y[32,D] += gelu(H)[32,FC] * W2T, reduce over FC ----
        const unsigned short* w2p =
            W2T + ((size_t)t * Dm + grp * 128 + nlane) * Fm + fc * FC + khalf;
        for (int kc = 0; kc < FC; kc += 32) {
            const unsigned short* gb = w2p + kc; // per-j stride = 16*Fm*2 bytes
            prefetch_wgp<0 * 65536 + 256>(gb);
            prefetch_wgp<1 * 65536 + 256>(gb);
            prefetch_wgp<2 * 65536 + 256>(gb);
            prefetch_wgp<3 * 65536 + 256>(gb);
            prefetch_wgp<4 * 65536 + 256>(gb);
            prefetch_wgp<5 * 65536 + 256>(gb);
            prefetch_wgp<6 * 65536 + 256>(gb);
            prefetch_wgp<7 * 65536 + 256>(gb);
            v16bf a = load_a(HlA, lane, kc, HSTR);
#pragma unroll
            for (int jj = 0; jj < 2; ++jj) { // two groups of 4 to cap pressure
                v16bf bfr[4];
#pragma unroll
                for (int j = 0; j < 4; ++j)
                    bfr[j] = load_b(gb + (size_t)((jj * 4 + j) * 16) * Fm);
#pragma unroll
                for (int j = 0; j < 4; ++j)
                    cy[jj * 4 + j] = __builtin_amdgcn_wmma_f32_16x16x32_bf16(
                        false, a, false, bfr[j], (short)0, cy[jj * 4 + j],
                        false, false);
            }
        }
    }

    // ---- epilogue: bias + scatter rows back to token positions ----
#pragma unroll
    for (int j = 0; j < 8; ++j) {
        const int nd   = grp * 128 + j * 16 + nlane;
        const float bias = b2[t * Dm + nd];
#pragma unroll
        for (int r = 0; r < 8; ++r) {
            const int m = mt * 16 + r + mbase;
            if (m < validRows) {
                const int tok = idxs[r0 + m];
                out[(size_t)tok * Dm + nd] = cy[j][r] + bias;
            }
        }
    }
}

// ---------------------------------------------------------------------------
extern "C" void kernel_launch(void* const* d_in, const int* in_sizes, int n_in,
                              void* d_out, int out_size, void* d_ws,
                              size_t ws_size, hipStream_t stream) {
    (void)in_sizes; (void)n_in; (void)out_size; (void)ws_size;

    const float* x  = (const float*)d_in[0];
    const int*   tt = (const int*)d_in[1];
    const float* W1 = (const float*)d_in[2];
    const float* b1 = (const float*)d_in[3];
    const float* W2 = (const float*)d_in[4];
    const float* b2 = (const float*)d_in[5];
    float* out = (float*)d_out;

    // workspace layout
    char* ws = (char*)d_ws;
    int* meta = (int*)ws;                                   // 9 ints, 256B slot
    int* idxs = (int*)(ws + 256);                           // (N+128) ints
    unsigned short* W1T =
        (unsigned short*)(ws + 256 + (size_t)(N_TOK + 128) * 4);
    unsigned short* W2T = W1T + (size_t)Tm * Fm * Dm;       // ~17 MB total

    build_groups<<<1, 1024, 0, stream>>>(tt, meta, idxs);

    const int convBlocks = (Tm * Fm * Dm) / 256; // 16384
    convert_w1<<<convBlocks, 256, 0, stream>>>(W1, W1T);
    convert_w2<<<convBlocks, 256, 0, stream>>>(W2, W2T);

    const int maxTiles = N_TOK / MT + Tm; // padded groups upper bound (1028)
    ffn_kernel<<<maxTiles, 256, 0, stream>>>(x, meta, idxs, W1T, b1, W2T, b2,
                                             out);
}
